// SimpleGRU_83992380440832
// MI455X (gfx1250) — matrix operand order repaired and checked by reading, not compiled, not run on hardware
//
#include <hip/hip_runtime.h>
#include <hip/hip_bf16.h>
#include <math.h>
#include <stdint.h>

typedef __bf16 bf16_t;
typedef __attribute__((ext_vector_type(16))) __bf16 v16bf;
typedef __attribute__((ext_vector_type(8)))  float  v8f;
typedef unsigned int u32;
typedef __attribute__((ext_vector_type(4))) u32 v4u;
typedef __attribute__((ext_vector_type(8))) int  v8i;
typedef __attribute__((ext_vector_type(4))) int  v4i;

#define GB    64
#define GT    512
#define GDIN  256
#define GH    1024
#define G3H   3072
#define GOUT  50257

// Tensor Data Mover path (clang-23 6-arg builtin); falls back to synchronous
// LDS staging if the builtin isn't available in this toolchain/pass.
#if defined(__HIP_DEVICE_COMPILE__) && \
    __has_builtin(__builtin_amdgcn_tensor_load_to_lds) && \
    __has_builtin(__builtin_amdgcn_s_wait_tensorcnt)
#define USE_TDM 1
#else
#define USE_TDM 0
#endif

union BF16Frag { v16bf v; uint4 q[2]; bf16_t e[16]; };

__device__ __forceinline__ v8f zero_v8f() {
  v8f v;
#pragma unroll
  for (int i = 0; i < 8; ++i) v[i] = 0.0f;
  return v;
}

__device__ __forceinline__ v8f wmma_bf16(v16bf a, v16bf b, v8f c) {
  // 8 args: (neg_a, A, neg_b, B, c_mod, C, reuse_a, reuse_b)
  return __builtin_amdgcn_wmma_f32_16x16x32_bf16(false, a, false, b, (short)0, c,
                                                 false, false);
}

#if USE_TDM
// Issue one TDM 2D tile load: tile0 x tile1 elements (2-byte), row stride
// `stride0` elements, from `gptr` into LDS byte offset `lds_off`.
__device__ __forceinline__ void tdm_load_2d(u32 lds_off, const void* gptr,
                                            u32 tile0, u32 tile1,
                                            u32 t0, u32 t1, u32 stride0) {
  const unsigned long long ga = (unsigned long long)(size_t)gptr;
  v4u g0;
  g0[0] = 1u;                                                // count=1, user D#
  g0[1] = lds_off;                                           // lds_addr
  g0[2] = (u32)ga;                                           // global_addr[31:0]
  g0[3] = (u32)((ga >> 32) & 0x01FFFFFFull) | (2u << 30);    // addr[56:32], type=2
  v8i g1;
  g1[0] = (int)(1u << 16);                                   // wg_mask=0, data_size=1 (2B)
  g1[1] = (int)((t0 & 0xFFFFu) << 16);                       // tensor_dim0[15:0]
  g1[2] = (int)((t0 >> 16) | ((t1 & 0xFFFFu) << 16));        // t0[31:16], t1[15:0]
  g1[3] = (int)((t1 >> 16) | (tile0 << 16));                 // t1[31:16], tile_dim0
  g1[4] = (int)tile1;                                        // tile_dim1 (tile_dim2=0)
  g1[5] = (int)stride0;                                      // tensor_dim0_stride[31:0]
  g1[6] = 0;
  g1[7] = 0;
  v4i gz4; gz4[0] = 0; gz4[1] = 0; gz4[2] = 0; gz4[3] = 0;   // 2D: groups 2/3 unused
  v8i gz8;
#pragma unroll
  for (int i = 0; i < 8; ++i) gz8[i] = 0;
  __builtin_amdgcn_tensor_load_to_lds(g0, g1, gz4, gz4, gz8, 0);
}
#endif

// ---------------------------------------------------------------------------
// Conversion / init kernels
// ---------------------------------------------------------------------------
__global__ __launch_bounds__(256) void cvt_x_kernel(const float* __restrict__ in,
                                                    bf16_t* __restrict__ out) {
  // out[t][b][d] = (bf16) in[b][t][d]
  int idx = blockIdx.x * 256 + threadIdx.x;
  if (idx >= GT * GB * GDIN) return;
  int t = idx / (GB * GDIN);
  int r = idx % (GB * GDIN);
  int b = r / GDIN;
  int d = r % GDIN;
  out[idx] = (bf16_t)in[(size_t)b * GT * GDIN + (size_t)t * GDIN + d];
}

__global__ __launch_bounds__(256) void cvt_kernel(const float* __restrict__ in,
                                                  bf16_t* __restrict__ out, int n) {
  int idx = blockIdx.x * 256 + threadIdx.x;
  if (idx < n) out[idx] = (bf16_t)in[idx];
}

__global__ __launch_bounds__(256) void zero_h_kernel(float* __restrict__ hf,
                                                     bf16_t* __restrict__ hb, int n) {
  int idx = blockIdx.x * 256 + threadIdx.x;
  if (idx < n) { hf[idx] = 0.0f; hb[idx] = (bf16_t)0.0f; }
}

// ---------------------------------------------------------------------------
// One GRU time step:  gates = x_t@Wih^T + h@Whh^T (+biases), fused nonlinearity.
// Grid: 64 WGs (16 hidden cols each) x 128 threads (4 waves, 16 rows each).
// h staged through LDS in double-buffered 64x128 chunks via the TDM.
// ---------------------------------------------------------------------------
__global__ __launch_bounds__(128) void gru_step_kernel(
    const bf16_t* __restrict__ xb,     // [64][256]  bf16, time slice t
    const bf16_t* __restrict__ hb_in,  // [64][1024] bf16
    const float*  __restrict__ h_in,   // [64][1024] f32
    const bf16_t* __restrict__ Wih,    // [3072][256]  bf16
    const bf16_t* __restrict__ Whh,    // [3072][1024] bf16
    const float*  __restrict__ bih,    // [3072]
    const float*  __restrict__ bhh,    // [3072]
    bf16_t* __restrict__ hb_out,       // [64][1024] bf16
    float*  __restrict__ h_out)        // [64][1024] f32
{
  __shared__ bf16_t sh[2][GB][128];    // 32 KB double buffer

  const int tid   = threadIdx.x;
  const int lane  = tid & 31;
  const int wave  = tid >> 5;
  const int c0    = blockIdx.x * 16;   // hidden-column tile base
  const int mbase = wave * 16;         // batch-row tile base

  const int mrow = lane & 15;
  const int kb   = (lane >> 4) * 16;   // A-frag K sub-base
  const int bcol = lane & 15;          // B-frag column within tile
  const int bk   = (lane >> 4) * 16;   // B-frag K group

#if USE_TDM
  u32 sh_off[2];
  sh_off[0] = (u32)(size_t)(&sh[0][0][0]);
  sh_off[1] = (u32)(size_t)(&sh[1][0][0]);
  if (wave == 0) {
    tdm_load_2d(sh_off[0], hb_in, 128u, (u32)GB, (u32)GH, (u32)GB, (u32)GH);
    __builtin_amdgcn_s_wait_tensorcnt(0);
  }
#else
  for (int i = tid; i < (GB * 128) / 8; i += 128)
    ((uint4*)sh[0])[i] =
        *(const uint4*)(hb_in + (size_t)(i >> 4) * GH + (i & 15) * 8);
#endif
  __syncthreads();

  v8f acc_r = zero_v8f();   // i_r + h_r
  v8f acc_z = zero_v8f();   // i_z + h_z
  v8f acc_i = zero_v8f();   // i_n only
  v8f acc_n = zero_v8f();   // h_n only

  // ---- input-projection part: K = 256 over x_t (L2-resident, global A) ----
#pragma unroll
  for (int kk = 0; kk < GDIN; kk += 32) {
    BF16Frag ua;
    const bf16_t* ap = xb + (size_t)(mbase + mrow) * GDIN + kk + kb;
    ua.q[0] = *(const uint4*)(ap);
    ua.q[1] = *(const uint4*)(ap + 8);
    const v16bf a = ua.v;

    const int col = c0 + bcol;
    v16bf wr = *(const v16bf*)(Wih + (size_t)(col)          * GDIN + kk + bk);
    v16bf wz = *(const v16bf*)(Wih + (size_t)(GH + col)     * GDIN + kk + bk);
    v16bf wn = *(const v16bf*)(Wih + (size_t)(2 * GH + col) * GDIN + kk + bk);
    acc_r = wmma_bf16(a, wr, acc_r);
    acc_z = wmma_bf16(a, wz, acc_z);
    acc_i = wmma_bf16(a, wn, acc_i);
  }

  // ---- recurrent part: K = 1024 over h, double-buffered 128-wide chunks ----
  for (int c = 0; c < 8; ++c) {
    const int cur = c & 1;
    const int nxt = cur ^ 1;
    const int kc0 = c * 128;

#if USE_TDM
    if (wave == 0 && c < 7) {
      // Kick next chunk's DMA before computing on the current one.
      tdm_load_2d(sh_off[nxt], hb_in + (kc0 + 128), 128u, (u32)GB,
                  (u32)GH, (u32)GB, (u32)GH);
    }
#endif

#pragma unroll
    for (int k4 = 0; k4 < 4; ++k4) {
      const int kk = k4 * 32;
      BF16Frag ua;
      const bf16_t* ap = &sh[cur][mbase + mrow][kk + kb];
      ua.q[0] = *(const uint4*)(ap);
      ua.q[1] = *(const uint4*)(ap + 8);
      const v16bf a = ua.v;

      const int col = c0 + bcol;
      const int kg  = kc0 + kk + bk;
      v16bf wr = *(const v16bf*)(Whh + (size_t)(col)          * GH + kg);
      v16bf wz = *(const v16bf*)(Whh + (size_t)(GH + col)     * GH + kg);
      v16bf wn = *(const v16bf*)(Whh + (size_t)(2 * GH + col) * GH + kg);
      acc_r = wmma_bf16(a, wr, acc_r);
      acc_z = wmma_bf16(a, wz, acc_z);
      acc_n = wmma_bf16(a, wn, acc_n);
    }

    if (c < 7) {
#if USE_TDM
      if (wave == 0) __builtin_amdgcn_s_wait_tensorcnt(0);
      __syncthreads();
#else
      __syncthreads();
      for (int i = tid; i < (GB * 128) / 8; i += 128)
        ((uint4*)sh[nxt])[i] =
            *(const uint4*)(hb_in + (size_t)(i >> 4) * GH + (kc0 + 128) +
                            (i & 15) * 8);
      __syncthreads();
#endif
    }
  }

  // ---- gate fusion + state update ----
  const int j   = c0 + (lane & 15);          // global hidden column
  const float br_ = bih[j] + bhh[j];
  const float bz_ = bih[GH + j] + bhh[GH + j];
  const float bi_ = bih[2 * GH + j];
  const float bh_ = bhh[2 * GH + j];
  const int rofs = (lane >> 4) * 8;

#pragma unroll
  for (int i = 0; i < 8; ++i) {
    const int row = mbase + i + rofs;        // global batch row
    const float hold = h_in[(size_t)row * GH + j];
    const float rv = 1.0f / (1.0f + expf(-(acc_r[i] + br_)));
    const float zv = 1.0f / (1.0f + expf(-(acc_z[i] + bz_)));
    const float nv = tanhf(acc_i[i] + bi_ + rv * (acc_n[i] + bh_));
    const float hv = (1.0f - zv) * nv + zv * hold;
    h_out[(size_t)row * GH + j]  = hv;
    hb_out[(size_t)row * GH + j] = (bf16_t)hv;
  }
}

// ---------------------------------------------------------------------------
// Dense head: logits[64][50257] = h @ W_dense^T + b. W_dense streamed fp32
// (206 MB, memory-bound @ 23.3 TB/s), converted to bf16 in registers so the
// pass stays bandwidth-bound instead of fp32-VALU-bound.
// ---------------------------------------------------------------------------
__global__ __launch_bounds__(128) void dense_kernel(
    const bf16_t* __restrict__ hb,   // [64][1024] bf16 final hidden
    const float*  __restrict__ Wd,   // [50257][1024] fp32
    const float*  __restrict__ bd,   // [50257]
    float* __restrict__ logits)      // [64][50257]
{
  const int tid   = threadIdx.x;
  const int lane  = tid & 31;
  const int wave  = tid >> 5;
  const int ncol0 = blockIdx.x * 64 + wave * 16;

  v8f acc[4];
#pragma unroll
  for (int m = 0; m < 4; ++m) acc[m] = zero_v8f();

  const int mrow = lane & 15;
  const int kb   = (lane >> 4) * 16;
  const int bcol = lane & 15;
  const int bk   = (lane >> 4) * 16;

  const int  wrow  = ncol0 + bcol;        // W_dense row = output column
  const bool valid = wrow < GOUT;
  const float* wbase = Wd + (size_t)(valid ? wrow : 0) * GH;

  for (int kk = 0; kk < GH; kk += 32) {
    // B fragment: 16 fp32 -> bf16 in registers
    BF16Frag ub;
    const float* wp = wbase + kk + bk;
    __builtin_prefetch((const void*)(wp + 128), 0, 0);
    float4 f0 = *(const float4*)(wp + 0);
    float4 f1 = *(const float4*)(wp + 4);
    float4 f2 = *(const float4*)(wp + 8);
    float4 f3 = *(const float4*)(wp + 12);
    ub.e[0]  = (bf16_t)f0.x; ub.e[1]  = (bf16_t)f0.y;
    ub.e[2]  = (bf16_t)f0.z; ub.e[3]  = (bf16_t)f0.w;
    ub.e[4]  = (bf16_t)f1.x; ub.e[5]  = (bf16_t)f1.y;
    ub.e[6]  = (bf16_t)f1.z; ub.e[7]  = (bf16_t)f1.w;
    ub.e[8]  = (bf16_t)f2.x; ub.e[9]  = (bf16_t)f2.y;
    ub.e[10] = (bf16_t)f2.z; ub.e[11] = (bf16_t)f2.w;
    ub.e[12] = (bf16_t)f3.x; ub.e[13] = (bf16_t)f3.y;
    ub.e[14] = (bf16_t)f3.z; ub.e[15] = (bf16_t)f3.w;
    const v16bf b = ub.v;

#pragma unroll
    for (int m = 0; m < 4; ++m) {
      BF16Frag ua;
      const bf16_t* ap = hb + (size_t)(m * 16 + mrow) * GH + kk + kb;
      ua.q[0] = *(const uint4*)(ap);
      ua.q[1] = *(const uint4*)(ap + 8);
      acc[m] = wmma_bf16(ua.v, b, acc[m]);
    }
  }

  const int col = ncol0 + (lane & 15);
  if (col < GOUT) {
    const float bb = bd[col];
    const int rofs = (lane >> 4) * 8;
#pragma unroll
    for (int m = 0; m < 4; ++m) {
#pragma unroll
      for (int i = 0; i < 8; ++i) {
        const int row = m * 16 + i + rofs;
        logits[(size_t)row * GOUT + col] = acc[m][i] + bb;
      }
    }
  }
}

// ---------------------------------------------------------------------------
// log_softmax over each of the 64 rows of logits
// ---------------------------------------------------------------------------
__global__ __launch_bounds__(256) void row_max_kernel(const float* __restrict__ logits,
                                                      float* __restrict__ rowmax) {
  __shared__ float red[256];
  const int b = blockIdx.x;
  float m = -3.4e38f;
  for (int i = threadIdx.x; i < GOUT; i += 256)
    m = fmaxf(m, logits[(size_t)b * GOUT + i]);
  red[threadIdx.x] = m;
  __syncthreads();
  for (int s = 128; s > 0; s >>= 1) {
    if (threadIdx.x < s) red[threadIdx.x] = fmaxf(red[threadIdx.x], red[threadIdx.x + s]);
    __syncthreads();
  }
  if (threadIdx.x == 0) rowmax[b] = red[0];
}

__global__ __launch_bounds__(256) void row_lse_kernel(const float* __restrict__ logits,
                                                      const float* __restrict__ rowmax,
                                                      float* __restrict__ rowlse) {
  __shared__ float red[256];
  const int b = blockIdx.x;
  const float mx = rowmax[b];
  float s = 0.0f;
  for (int i = threadIdx.x; i < GOUT; i += 256)
    s += expf(logits[(size_t)b * GOUT + i] - mx);
  red[threadIdx.x] = s;
  __syncthreads();
  for (int st = 128; st > 0; st >>= 1) {
    if (threadIdx.x < st) red[threadIdx.x] += red[threadIdx.x + st];
    __syncthreads();
  }
  if (threadIdx.x == 0) rowlse[b] = logf(red[0]);
}

__global__ __launch_bounds__(256) void logsm_kernel(const float* __restrict__ logits,
                                                    const float* __restrict__ rowmax,
                                                    const float* __restrict__ rowlse,
                                                    float* __restrict__ out) {
  int idx = blockIdx.x * 256 + threadIdx.x;
  if (idx >= GB * GOUT) return;
  const int b = idx / GOUT;
  out[idx] = logits[idx] - rowmax[b] - rowlse[b];
}

// ---------------------------------------------------------------------------
// Host launcher
// ---------------------------------------------------------------------------
extern "C" void kernel_launch(void* const* d_in, const int* in_sizes, int n_in,
                              void* d_out, int out_size, void* d_ws, size_t ws_size,
                              hipStream_t stream) {
  (void)in_sizes; (void)n_in; (void)out_size; (void)ws_size;
  const float* in_seq  = (const float*)d_in[0];
  const float* W_ih    = (const float*)d_in[1];
  const float* W_hh    = (const float*)d_in[2];
  const float* b_ih    = (const float*)d_in[3];
  const float* b_hh    = (const float*)d_in[4];
  const float* W_dense = (const float*)d_in[5];
  const float* b_dense = (const float*)d_in[6];
  float* out = (float*)d_out;

  // Workspace layout (all chunks are multiples of 256 B) -- ~38.4 MB total
  char* p = (char*)d_ws;
  bf16_t* Xb    = (bf16_t*)p; p += (size_t)GT * GB * GDIN * 2;   // [T][B][D] bf16
  bf16_t* Wih_b = (bf16_t*)p; p += (size_t)G3H * GDIN * 2;
  bf16_t* Whh_b = (bf16_t*)p; p += (size_t)G3H * GH * 2;
  bf16_t* hb0   = (bf16_t*)p; p += (size_t)GB * GH * 2;
  bf16_t* hb1   = (bf16_t*)p; p += (size_t)GB * GH * 2;
  float*  hf0   = (float*)p;  p += (size_t)GB * GH * 4;
  float*  hf1   = (float*)p;  p += (size_t)GB * GH * 4;
  float*  logits= (float*)p;  p += (size_t)GB * GOUT * 4;
  float*  rowmax= (float*)p;  p += 256;
  float*  rowlse= (float*)p;  p += 256;

  bf16_t* hb[2] = {hb0, hb1};
  float*  hf[2] = {hf0, hf1};

  // 1) Precision conversion (fp32 -> bf16), x transposed to [T][B][D]
  {
    int n = GT * GB * GDIN;
    cvt_x_kernel<<<(n + 255) / 256, 256, 0, stream>>>(in_seq, Xb);
  }
  cvt_kernel<<<(G3H * GDIN + 255) / 256, 256, 0, stream>>>(W_ih, Wih_b, G3H * GDIN);
  cvt_kernel<<<(G3H * GH + 255) / 256, 256, 0, stream>>>(W_hh, Whh_b, G3H * GH);
  zero_h_kernel<<<(GB * GH + 255) / 256, 256, 0, stream>>>(hf0, hb0, GB * GH);

  // 2) Sequential GRU recurrence: 512 step kernels, ping-pong h buffers
  for (int t = 0; t < GT; ++t) {
    const bf16_t* xt = Xb + (size_t)t * GB * GDIN;
    const int r = t & 1, w = (t + 1) & 1;
    gru_step_kernel<<<GH / 16, 128, 0, stream>>>(
        xt, hb[r], hf[r], Wih_b, Whh_b, b_ih, b_hh, hb[w], hf[w]);
  }
  // after 512 steps (even), final hidden state is in buffer 0

  // 3) Dense head (WMMA, fp32 weights converted in-register)
  dense_kernel<<<(GOUT + 63) / 64, 128, 0, stream>>>(hb[0], W_dense, b_dense, logits);

  // 4) log_softmax
  row_max_kernel<<<GB, 256, 0, stream>>>(logits, rowmax);
  row_lse_kernel<<<GB, 256, 0, stream>>>(logits, rowmax, rowlse);
  {
    int n = GB * GOUT;
    logsm_kernel<<<(n + 255) / 256, 256, 0, stream>>>(logits, rowmax, rowlse, out);
  }
}
